// CausalSelfAttention_44530220925508
// MI455X (gfx1250) — matrix operand order repaired and checked by reading, not compiled, not run on hardware
//
#include <hip/hip_runtime.h>
#include <hip/hip_bf16.h>

typedef __bf16 bf16;
typedef bf16 v16bf __attribute__((ext_vector_type(16)));
typedef bf16 v8bf  __attribute__((ext_vector_type(8)));
typedef bf16 v4bf  __attribute__((ext_vector_type(4)));
typedef float v8f  __attribute__((ext_vector_type(8)));

static constexpr int Bc  = 4;
static constexpr int Sc  = 2048;
static constexpr int Hc  = 1024;
static constexpr int NHc = 16;
static constexpr int HDc = 64;

__device__ __forceinline__ v16bf cat8(v8bf lo, v8bf hi) {
  return __builtin_shufflevector(lo, hi, 0,1,2,3,4,5,6,7,8,9,10,11,12,13,14,15);
}

__device__ __forceinline__ v8f wmma_bf16(v16bf a, v16bf b, v8f c) {
  // D = A(16x32 bf16) * B(32x16 bf16) + C(16x16 f32)
  return __builtin_amdgcn_wmma_f32_16x16x32_bf16(false, a, false, b, (short)0, c,
                                                 false, false);
}

// Async global->LDS copy of 16 bytes (ASYNCcnt-tracked, no VGPR round trip).
__device__ __forceinline__ void async_copy_b128(const void* gsrc, void* lds_dst) {
  unsigned lds_off = (unsigned)(uintptr_t)lds_dst;   // flat addr low 32b = LDS byte addr
  asm volatile("global_load_async_to_lds_b128 %0, %1, off"
               :: "v"(lds_off), "v"(gsrc) : "memory");
}
__device__ __forceinline__ void wait_asynccnt0() {
  asm volatile("s_wait_asynccnt 0x0" ::: "memory");
}

// ---------------------------------------------------------------------------
// fp32 -> bf16 conversion (vectorized: float4 in, 8-byte bf16x4 out)
// ---------------------------------------------------------------------------
__global__ void f32_to_bf16_kernel(const float* __restrict__ src,
                                   bf16* __restrict__ dst, int n4) {
  int i = blockIdx.x * blockDim.x + threadIdx.x;
  if (i < n4) {
    float4 f = ((const float4*)src)[i];
    v4bf o;
    o[0] = (bf16)f.x; o[1] = (bf16)f.y; o[2] = (bf16)f.z; o[3] = (bf16)f.w;
    ((v4bf*)dst)[i] = o;
  }
}

// ---------------------------------------------------------------------------
// GEMM: Out[m,n] = sum_k A[m,k] * Bw[n,k] + bias[n]   (A: MxK, Bw: NxK, bf16)
// Block tile 128x128, K-step 32, double-buffered LDS (one barrier per step).
// 256 threads = 8 waves in 4(M) x 2(N); each wave = 32x64 = 2x4 WMMA tiles.
// ---------------------------------------------------------------------------
template<bool OUT_BF16>
__global__ __launch_bounds__(256) void gemm_bt_bias_kernel(
    const bf16* __restrict__ A, const bf16* __restrict__ Bw,
    const float* __restrict__ bias, void* __restrict__ OutP,
    int M, int N, int K) {
  constexpr int BM = 128, BN = 128, BK = 32;
  __shared__ alignas(16) bf16 As[2][BM * BK];
  __shared__ alignas(16) bf16 Bs[2][BN * BK];

  const int tid  = threadIdx.x;
  const int wave = tid >> 5, lane = tid & 31;
  const int waveM = wave & 3, waveN = wave >> 2;
  const int half = lane >> 4, ln = lane & 15;
  const int m0 = blockIdx.y * BM, n0 = blockIdx.x * BN;

  // Per-thread 16B staging chunks: 512 chunks per 128x32 tile, 2 per thread.
  int rowc[2], colc[2];
  #pragma unroll
  for (int i = 0; i < 2; ++i) {
    int c = tid + i * 256;
    rowc[i] = c >> 2;
    colc[i] = (c & 3) << 3;   // bf16 units
  }

  v8f acc[2][4];
  #pragma unroll
  for (int mi = 0; mi < 2; ++mi)
    #pragma unroll
    for (int ni = 0; ni < 4; ++ni)
      acc[mi][ni] = (v8f)0.0f;

  // Preload tile 0 into LDS buffer 0.
  uint4 pa[2], pb[2];
  #pragma unroll
  for (int i = 0; i < 2; ++i) {
    pa[i] = *(const uint4*)&A[(size_t)(m0 + rowc[i]) * K + colc[i]];
    pb[i] = *(const uint4*)&Bw[(size_t)(n0 + rowc[i]) * K + colc[i]];
  }
  #pragma unroll
  for (int i = 0; i < 2; ++i) {
    *(uint4*)&As[0][rowc[i] * BK + colc[i]] = pa[i];
    *(uint4*)&Bs[0][rowc[i] * BK + colc[i]] = pb[i];
  }
  __syncthreads();

  const int nk = K / BK;
  for (int kt = 0; kt < nk; ++kt) {
    const int cur = kt & 1;
    // Prefetch next tile from global into registers (overlaps with WMMA).
    if (kt + 1 < nk) {
      const int k0 = (kt + 1) * BK;
      #pragma unroll
      for (int i = 0; i < 2; ++i) {
        pa[i] = *(const uint4*)&A[(size_t)(m0 + rowc[i]) * K + k0 + colc[i]];
        pb[i] = *(const uint4*)&Bw[(size_t)(n0 + rowc[i]) * K + k0 + colc[i]];
      }
    }
    // L2 prefetch two tiles ahead.
    if (kt + 2 < nk) {
      const int k2 = (kt + 2) * BK;
      __builtin_prefetch(&A[(size_t)(m0 + rowc[0]) * K + k2 + colc[0]], 0, 0);
      __builtin_prefetch(&Bw[(size_t)(n0 + rowc[0]) * K + k2 + colc[0]], 0, 0);
    }

    // Fragments from LDS buffer `cur`, then the 8 WMMAs.
    v16bf af[2];
    #pragma unroll
    for (int mi = 0; mi < 2; ++mi) {
      const bf16* p = &As[cur][(waveM * 32 + mi * 16 + ln) * BK];
      af[mi] = cat8(*(const v8bf*)(p + half * 8),
                    *(const v8bf*)(p + 16 + half * 8));
    }
    v16bf bfArr[4];
    #pragma unroll
    for (int ni = 0; ni < 4; ++ni) {
      const bf16* p = &Bs[cur][(waveN * 64 + ni * 16 + ln) * BK];
      bfArr[ni] = cat8(*(const v8bf*)(p + half * 8),
                       *(const v8bf*)(p + 16 + half * 8));
    }
    #pragma unroll
    for (int mi = 0; mi < 2; ++mi)
      #pragma unroll
      for (int ni = 0; ni < 4; ++ni)
        acc[mi][ni] = wmma_bf16(af[mi], bfArr[ni], acc[mi][ni]);

    // Stage next tile into the other LDS buffer.
    if (kt + 1 < nk) {
      #pragma unroll
      for (int i = 0; i < 2; ++i) {
        *(uint4*)&As[cur ^ 1][rowc[i] * BK + colc[i]] = pa[i];
        *(uint4*)&Bs[cur ^ 1][rowc[i] * BK + colc[i]] = pb[i];
      }
    }
    __syncthreads();
  }

  // Epilogue: C layout -> element r is row (half*8 + r), column ln.
  #pragma unroll
  for (int mi = 0; mi < 2; ++mi) {
    #pragma unroll
    for (int ni = 0; ni < 4; ++ni) {
      int gn = n0 + waveN * 64 + ni * 16 + ln;
      float bv = bias[gn];
      #pragma unroll
      for (int r = 0; r < 8; ++r) {
        int gm = m0 + waveM * 32 + mi * 16 + half * 8 + r;
        float v = acc[mi][ni][r] + bv;
        if constexpr (OUT_BF16)
          ((bf16*)OutP)[(size_t)gm * N + gn] = (bf16)v;
        else
          ((float*)OutP)[(size_t)gm * N + gn] = v;
      }
    }
  }
}

// ---------------------------------------------------------------------------
// Flash attention, causal. Grid (S/64, NH, B), 128 threads = 4 waves.
// K tile staged with GLOBAL_LOAD_ASYNC_TO_LDS_B128 (ASYNCcnt path); V tile is
// transposed through VGPRs so PV B-fragments are two 16B LDS loads each.
// ---------------------------------------------------------------------------
__global__ __launch_bounds__(128) void flash_attn_kernel(
    const bf16* __restrict__ Q, const bf16* __restrict__ Kg,
    const bf16* __restrict__ Vg, bf16* __restrict__ Ctx, float scale) {
  constexpr int S = Sc, H = Hc, HD = HDc;
  __shared__ alignas(16) bf16 Ks[64 * 64];      // key-major
  __shared__ alignas(16) bf16 Vts[64 * 64];     // transposed: Vts[hd*64 + key]
  __shared__ alignas(16) bf16 Ps[4][16 * 64];   // per-wave P staging

  const int tid  = threadIdx.x;
  const int wave = tid >> 5, lane = tid & 31;
  const int half = lane >> 4, ln = lane & 15;
  const int qt = blockIdx.x, head = blockIdx.y, b = blockIdx.z;
  const int q0 = qt * 64 + wave * 16;
  const size_t rowBase = ((size_t)b * S) * H + (size_t)head * HD;

  // Q fragments: 16 rows x 64 HD = two 16x32 A fragments, kept in VGPRs.
  v16bf qf[2];
  {
    const bf16* qp = Q + rowBase + (size_t)(q0 + ln) * H;
    #pragma unroll
    for (int c = 0; c < 2; ++c)
      qf[c] = cat8(*(const v8bf*)(qp + c * 32 + half * 8),
                   *(const v8bf*)(qp + c * 32 + 16 + half * 8));
  }

  v8f o[4];
  #pragma unroll
  for (int n = 0; n < 4; ++n) o[n] = (v8f)0.0f;
  float mrow[8], lrow[8];
  #pragma unroll
  for (int r = 0; r < 8; ++r) { mrow[r] = -1e30f; lrow[r] = 0.0f; }

  for (int kt = 0; kt <= qt; ++kt) {
    const int kb = kt * 64;
    // Stage K (async, no VGPR round trip) and V (transposed) into LDS.
    #pragma unroll
    for (int i = 0; i < 4; ++i) {
      int c = tid + i * 128;            // 0..511 16-byte chunks
      int row = c >> 3;
      int col8 = (c & 7) << 3;
      async_copy_b128(Kg + rowBase + (size_t)(kb + row) * H + col8,
                      &Ks[row * 64 + col8]);
      v8bf vv = *(const v8bf*)(Vg + rowBase + (size_t)(kb + row) * H + col8);
      #pragma unroll
      for (int j = 0; j < 8; ++j) Vts[(col8 + j) * 64 + row] = vv[j];
    }
    wait_asynccnt0();
    __syncthreads();

    // S = Q K^T : load all 4 B-fragments per k-chunk, then batch the WMMAs.
    v8f sacc[4];
    #pragma unroll
    for (int n = 0; n < 4; ++n) sacc[n] = (v8f)0.0f;
    #pragma unroll
    for (int c = 0; c < 2; ++c) {
      v16bf kf[4];
      #pragma unroll
      for (int n = 0; n < 4; ++n)
        kf[n] = cat8(*(const v8bf*)&Ks[(n * 16 + ln) * 64 + c * 32 + half * 8],
                     *(const v8bf*)&Ks[(n * 16 + ln) * 64 + c * 32 + 16 + half * 8]);
      #pragma unroll
      for (int n = 0; n < 4; ++n)
        sacc[n] = wmma_bf16(qf[c], kf[n], sacc[n]);
    }

    // Scale + causal mask (C layout: row = half*8+r, key col = kb + n*16+ln)
    float sv[4][8];
    #pragma unroll
    for (int n = 0; n < 4; ++n)
      #pragma unroll
      for (int r = 0; r < 8; ++r) {
        int kgi = kb + n * 16 + ln;
        int qgi = q0 + half * 8 + r;
        float s = sacc[n][r] * scale;
        sv[n][r] = (kgi > qgi) ? -1e30f : s;
      }

    // Online softmax: reductions within the 16-lane half owning each row.
    #pragma unroll
    for (int r = 0; r < 8; ++r) {
      float t = fmaxf(fmaxf(sv[0][r], sv[1][r]), fmaxf(sv[2][r], sv[3][r]));
      #pragma unroll
      for (int off = 1; off < 16; off <<= 1)
        t = fmaxf(t, __shfl_xor(t, off, 32));
      float mn = fmaxf(mrow[r], t);
      float rsum = 0.0f;
      #pragma unroll
      for (int n = 0; n < 4; ++n) {
        float p = __expf(sv[n][r] - mn);
        sv[n][r] = p;
        rsum += p;
      }
      #pragma unroll
      for (int off = 1; off < 16; off <<= 1)
        rsum += __shfl_xor(rsum, off, 32);
      float fac = __expf(mrow[r] - mn);
      #pragma unroll
      for (int n = 0; n < 4; ++n) o[n][r] *= fac;
      lrow[r] = lrow[r] * fac + rsum;
      mrow[r] = mn;
    }

    // P: C layout -> A-fragment layout via per-wave LDS staging (bf16).
    #pragma unroll
    for (int n = 0; n < 4; ++n)
      #pragma unroll
      for (int r = 0; r < 8; ++r)
        Ps[wave][(half * 8 + r) * 64 + n * 16 + ln] = (bf16)sv[n][r];

    // O += P V : A frag from Ps, B frags from transposed V (two 16B loads each).
    #pragma unroll
    for (int c = 0; c < 2; ++c) {
      const bf16* pp = &Ps[wave][ln * 64 + c * 32];
      v16bf pf = cat8(*(const v8bf*)(pp + half * 8),
                      *(const v8bf*)(pp + 16 + half * 8));
      v16bf vf[4];
      #pragma unroll
      for (int n = 0; n < 4; ++n)
        vf[n] = cat8(*(const v8bf*)&Vts[(n * 16 + ln) * 64 + c * 32 + half * 8],
                     *(const v8bf*)&Vts[(n * 16 + ln) * 64 + c * 32 + 16 + half * 8]);
      #pragma unroll
      for (int n = 0; n < 4; ++n)
        o[n] = wmma_bf16(pf, vf[n], o[n]);
    }
    __syncthreads();
  }

  // Normalize and store context (bf16) back in (B*S, H) layout.
  #pragma unroll
  for (int r = 0; r < 8; ++r) {
    float inv = 1.0f / lrow[r];
    int qgi = q0 + half * 8 + r;
    bf16* outp = Ctx + rowBase + (size_t)qgi * H;
    #pragma unroll
    for (int n = 0; n < 4; ++n)
      outp[n * 16 + ln] = (bf16)(o[n][r] * inv);
  }
}

// ---------------------------------------------------------------------------
extern "C" void kernel_launch(void* const* d_in, const int* in_sizes, int n_in,
                              void* d_out, int out_size, void* d_ws,
                              size_t ws_size, hipStream_t stream) {
  (void)in_sizes; (void)n_in; (void)out_size; (void)ws_size;
  const float* x  = (const float*)d_in[0];
  const float* Wq = (const float*)d_in[1];
  const float* bq = (const float*)d_in[2];
  const float* Wk = (const float*)d_in[3];
  const float* bk = (const float*)d_in[4];
  const float* Wv = (const float*)d_in[5];
  const float* bv = (const float*)d_in[6];
  const float* Wo = (const float*)d_in[7];
  const float* bo = (const float*)d_in[8];
  float* out = (float*)d_out;

  const size_t MS = (size_t)Bc * Sc;          // 8192 rows
  const size_t HH = (size_t)Hc * Hc;
  bf16* xb  = (bf16*)d_ws;                    // x in bf16
  bf16* wqb = xb + MS * Hc;
  bf16* wkb = wqb + HH;
  bf16* wvb = wkb + HH;
  bf16* wob = wvb + HH;
  bf16* qb  = wob + HH;                       // Q/K/V projections (bf16)
  bf16* kbuf = qb + MS * Hc;
  bf16* vbuf = kbuf + MS * Hc;
  bf16* cbuf = vbuf + MS * Hc;                // attention context (bf16)

  auto cvt = [&](const float* s, bf16* d, size_t n) {
    int n4 = (int)(n / 4);
    f32_to_bf16_kernel<<<(n4 + 255) / 256, 256, 0, stream>>>(s, d, n4);
  };
  cvt(x,  xb,  MS * Hc);
  cvt(Wq, wqb, HH);
  cvt(Wk, wkb, HH);
  cvt(Wv, wvb, HH);
  cvt(Wo, wob, HH);

  dim3 gg(Hc / 128, (unsigned)(MS / 128));
  gemm_bt_bias_kernel<true><<<gg, 256, 0, stream>>>(xb, wqb, bq, qb,
                                                    (int)MS, Hc, Hc);
  gemm_bt_bias_kernel<true><<<gg, 256, 0, stream>>>(xb, wkb, bk, kbuf,
                                                    (int)MS, Hc, Hc);
  gemm_bt_bias_kernel<true><<<gg, 256, 0, stream>>>(xb, wvb, bv, vbuf,
                                                    (int)MS, Hc, Hc);

  dim3 fg(Sc / 64, NHc, Bc);
  flash_attn_kernel<<<fg, 128, 0, stream>>>(qb, kbuf, vbuf, cbuf, 0.125f);

  gemm_bt_bias_kernel<false><<<gg, 256, 0, stream>>>(cbuf, wob, bo, out,
                                                     (int)MS, Hc, Hc);
}
